// Mix8Net_52553219834296
// MI455X (gfx1250) — compile-verified, matrix operand order checked by reading, not run on hardware
//
#include <hip/hip_runtime.h>
#include <hip/hip_bf16.h>

// ---------------------------------------------------------------------------
// Mix8Net on gfx1250: every layer is a [Npix x 128] x [128 x 128] GEMM done
// with v_wmma_f32_16x16x32_bf16 (f32 accumulate). Directions processed
// sequentially (workspace = cur + tmp bf16 buffers + packed weights).
// - Weights staged per-workgroup into LDS (16 waves share each 32KB B matrix)
// - Fragments stored split-half (lo 16B / hi 16B planes) so ds_load_b128 has
//   16B lane stride -> conflict-free (2-cycle) LDS reads
// - M-tile = 32 pixels per wave: each B fragment feeds 2 WMMAs -> 1 ds_load
//   per WMMA
// ---------------------------------------------------------------------------

typedef __attribute__((ext_vector_type(8)))  __bf16 v8bf;
typedef __attribute__((ext_vector_type(16))) __bf16 v16bf;
typedef __attribute__((ext_vector_type(8)))  float  v8f;

#define NPIX   230400      // 1024 * 15 * 15
#define MCH    128
#define PTILES 14400       // NPIX / 16
#define GBLK   450         // (NPIX/32) / 16 waves per block

__device__ __forceinline__ float silu_f(float v) {
  return v / (1.0f + __expf(-v));
}

__device__ __forceinline__ v16bf cat16(v8bf lo, v8bf hi) {
  return __builtin_shufflevector(lo, hi, 0,1,2,3,4,5,6,7,8,9,10,11,12,13,14,15);
}

__device__ __forceinline__ v16bf zero16() {
  v16bf z;
  #pragma unroll
  for (int i = 0; i < 16; ++i) z[i] = (__bf16)0.0f;
  return z;
}

// A-fragment (16 pixels x 32 K, bf16) straight from a pixel-major row.
__device__ __forceinline__ v16bf load_a(const __bf16* __restrict__ row, int c0) {
  v8bf lo = *(const v8bf*)(row + c0);
  v8bf hi = *(const v8bf*)(row + c0 + 16);
  return cat16(lo, hi);
}

// B-fragment from LDS, split-half layout: frag = 64 chunks of 16B,
// chunk[lane] = lane's lo half, chunk[32+lane] = lane's hi half.
__device__ __forceinline__ v16bf load_b_lds(const v8bf* __restrict__ lbs, int frag, int lane) {
  v8bf lo = lbs[frag * 64 + lane];
  v8bf hi = lbs[frag * 64 + 32 + lane];
  return cat16(lo, hi);
}

// ---------------------------------------------------------------------------
// Weight pre-pack: src = nMat matrices of [N][K] fp32 (conv weight w[o][c],
// i.e. B[k=c][n=o] = w[n][k]).  Fragment order: frag = (mat, kt, nt),
// 512 bf16 per fragment in split-half plane order (lo halves then hi halves).
// ---------------------------------------------------------------------------
__global__ void prep_kernel(const float* __restrict__ src, __bf16* __restrict__ dst,
                            int nMat, int N, int K) {
  int t = blockIdx.x * blockDim.x + threadIdx.x;
  int ntiles = N >> 4, ktiles = K >> 5;
  int total = nMat * ktiles * ntiles * 512;
  if (t >= total) return;
  int h    = t & 15;
  int lane = (t >> 4) & 31;
  int frag = t >> 9;
  int nt   = frag % ntiles;
  int km   = frag / ntiles;
  int kt   = km % ktiles;
  int mat  = km / ktiles;
  int n    = nt * 16 + (lane & 15);
  int base = (lane & 16) ? 8 : 0;
  int k    = kt * 32 + base + (h & 7) + ((h & 8) ? 16 : 0);
  // split-half layout: lo plane (h<8) at +0, hi plane (h>=8) at +256 elems
  int dsti = frag * 512 + ((h & 8) << 5) + lane * 8 + (h & 7);
  dst[dsti] = (__bf16)src[((size_t)mat * N + n) * K + k];
}

// ---------------------------------------------------------------------------
// Initial directional conv: Cin=2, 3 taps -> 128 ch, SiLU, bf16 out. Scalar.
// ---------------------------------------------------------------------------
__global__ __launch_bounds__(256) void init_dirconv_kernel(
    const float* __restrict__ xin, const float* __restrict__ w,
    const float* __restrict__ bias, __bf16* __restrict__ out,
    int dy0, int dx0, int dy2, int dx2) {
  int t = blockIdx.x * blockDim.x + threadIdx.x;
  if (t >= NPIX * MCH) return;
  int o = t & 127;
  int p = t >> 7;
  int b = p / 225, rem = p % 225;
  int y = rem / 15, xx = rem % 15;
  int dys[3] = {dy0, 0, dy2};
  int dxs[3] = {dx0, 0, dx2};
  float acc = bias[o];
  #pragma unroll
  for (int tap = 0; tap < 3; ++tap) {
    int yy = y + dys[tap], xc = xx + dxs[tap];
    if ((unsigned)yy < 15u && (unsigned)xc < 15u) {
      const float* xb = xin + ((b * 2 + 0) * 15 + yy) * 15 + xc;
      acc += w[tap * 256 + o * 2 + 0] * xb[0];
      acc += w[tap * 256 + o * 2 + 1] * xb[225];   // channel stride = 15*15
    }
  }
  out[(size_t)p * MCH + o] = (__bf16)silu_f(acc);
}

// ---------------------------------------------------------------------------
// Directional 3-tap conv: one wave = 32 pixels x 128 outs, 16 waves/block.
// Tap weights (32KB each) staged in LDS, shared by all 16 waves.
// 3 taps x 4 kt x 8 nt x 2 tiles = 192 WMMAs/wave, 1 ds_load_b128 per WMMA.
// ---------------------------------------------------------------------------
__global__ __launch_bounds__(512) void dirconv_kernel(
    const __bf16* __restrict__ in, __bf16* __restrict__ out,
    const __bf16* __restrict__ wf, const float* __restrict__ bias,
    int dy0, int dx0, int dy2, int dx2) {
  __shared__ v8bf lbs[32 * 64];                // 32 frags * 1KB = 32KB
  int lane = threadIdx.x & 31;
  int wave = blockIdx.x * 16 + (threadIdx.x >> 5);
  int row0 = wave * 32;
  int c0base = (lane & 16) ? 8 : 0;

  const __bf16 *p0[2], *pC[2], *p2[2];
  bool ok0[2], ok2[2];
  #pragma unroll
  for (int t2 = 0; t2 < 2; ++t2) {
    int mr = row0 + t2 * 16 + (lane & 15);
    int b = mr / 225, rem = mr % 225;
    int y = rem / 15, x = rem % 15;
    int y0 = y + dy0, x0 = x + dx0;
    int y2 = y + dy2, x2 = x + dx2;
    ok0[t2] = ((unsigned)y0 < 15u) && ((unsigned)x0 < 15u);
    ok2[t2] = ((unsigned)y2 < 15u) && ((unsigned)x2 < 15u);
    pC[t2] = in + (size_t)mr * MCH;
    p0[t2] = in + ((size_t)b * 225 + y0 * 15 + x0) * MCH;
    p2[t2] = in + ((size_t)b * 225 + y2 * 15 + x2) * MCH;
  }

  v8f acc[2][8];
  #pragma unroll
  for (int t2 = 0; t2 < 2; ++t2)
    #pragma unroll
    for (int i = 0; i < 8; ++i)
      #pragma unroll
      for (int j = 0; j < 8; ++j) acc[t2][i][j] = 0.0f;

  #pragma unroll
  for (int tap = 0; tap < 3; ++tap) {
    __syncthreads();                           // previous tap's reads done
    {
      const v8bf* src = (const v8bf*)(wf + (size_t)tap * 32 * 512);
      #pragma unroll
      for (int i = 0; i < 4; ++i) {
        int idx = i * 512 + threadIdx.x;       // 2048 chunks of 16B
        lbs[idx] = src[idx];
      }
    }
    __syncthreads();

    #pragma unroll
    for (int kt = 0; kt < 4; ++kt) {
      v16bf a[2];
      #pragma unroll
      for (int t2 = 0; t2 < 2; ++t2) {
        const __bf16* rT = (tap == 0) ? p0[t2] : ((tap == 1) ? pC[t2] : p2[t2]);
        bool okT = (tap == 0) ? ok0[t2] : ((tap == 1) ? true : ok2[t2]);
        a[t2] = zero16();
        if (okT) a[t2] = load_a(rT, kt * 32 + c0base);
      }
      #pragma unroll
      for (int nt = 0; nt < 8; ++nt) {
        v16bf bb = load_b_lds(lbs, kt * 8 + nt, lane);
        acc[0][nt] = __builtin_amdgcn_wmma_f32_16x16x32_bf16(false, a[0], false, bb, (short)0, acc[0][nt], false, false);
        acc[1][nt] = __builtin_amdgcn_wmma_f32_16x16x32_bf16(false, a[1], false, bb, (short)0, acc[1][nt], false, false);
      }
    }
  }

  int col = lane & 15;
  int radd = (lane & 16) ? 8 : 0;
  #pragma unroll
  for (int t2 = 0; t2 < 2; ++t2)
    #pragma unroll
    for (int nt = 0; nt < 8; ++nt) {
      int ch = nt * 16 + col;
      float bn = bias[ch];
      #pragma unroll
      for (int r = 0; r < 8; ++r) {
        float v = silu_f(acc[t2][nt][r] + bn);
        int pr = row0 + t2 * 16 + r + radd;
        out[(size_t)pr * MCH + ch] = (__bf16)v;
      }
    }
}

// ---------------------------------------------------------------------------
// 1x1 conv GEMM: optional SiLU + optional residual add.
// One wave = 32 pixels x 128 outs; 64 WMMAs/wave, 1 ds_load per WMMA.
// ---------------------------------------------------------------------------
__global__ __launch_bounds__(512) void gemm1x1_kernel(
    const __bf16* __restrict__ in, __bf16* __restrict__ out,
    const __bf16* __restrict__ wf, const float* __restrict__ bias,
    const __bf16* __restrict__ res, int do_silu) {
  __shared__ v8bf lbs[32 * 64];                // 32KB
  int lane = threadIdx.x & 31;
  int wave = blockIdx.x * 16 + (threadIdx.x >> 5);
  int row0 = wave * 32;
  int c0base = (lane & 16) ? 8 : 0;
  const __bf16* arow[2];
  arow[0] = in + (size_t)(row0 + (lane & 15)) * MCH;
  arow[1] = arow[0] + 16 * MCH;

  {
    const v8bf* src = (const v8bf*)wf;
    #pragma unroll
    for (int i = 0; i < 4; ++i) {
      int idx = i * 512 + threadIdx.x;
      lbs[idx] = src[idx];
    }
  }
  __syncthreads();

  v8f acc[2][8];
  #pragma unroll
  for (int t2 = 0; t2 < 2; ++t2)
    #pragma unroll
    for (int i = 0; i < 8; ++i)
      #pragma unroll
      for (int j = 0; j < 8; ++j) acc[t2][i][j] = 0.0f;

  #pragma unroll
  for (int kt = 0; kt < 4; ++kt) {
    v16bf a0 = load_a(arow[0], kt * 32 + c0base);
    v16bf a1 = load_a(arow[1], kt * 32 + c0base);
    #pragma unroll
    for (int nt = 0; nt < 8; ++nt) {
      v16bf bb = load_b_lds(lbs, kt * 8 + nt, lane);
      acc[0][nt] = __builtin_amdgcn_wmma_f32_16x16x32_bf16(false, a0, false, bb, (short)0, acc[0][nt], false, false);
      acc[1][nt] = __builtin_amdgcn_wmma_f32_16x16x32_bf16(false, a1, false, bb, (short)0, acc[1][nt], false, false);
    }
  }

  int col = lane & 15;
  int radd = (lane & 16) ? 8 : 0;
  #pragma unroll
  for (int t2 = 0; t2 < 2; ++t2)
    #pragma unroll
    for (int nt = 0; nt < 8; ++nt) {
      int ch = nt * 16 + col;
      float bn = bias[ch];
      #pragma unroll
      for (int r = 0; r < 8; ++r) {
        float v = acc[t2][nt][r] + bn;
        if (do_silu) v = silu_f(v);
        int pr = row0 + t2 * 16 + r + radd;
        if (res) v += (float)res[(size_t)pr * MCH + ch];
        out[(size_t)pr * MCH + ch] = (__bf16)v;
      }
    }
}

// ---------------------------------------------------------------------------
// Final 1x1 conv: 128 -> 64, f32 output scattered into [B,4,64,15,15].
// ---------------------------------------------------------------------------
__global__ __launch_bounds__(512) void final_kernel(
    const __bf16* __restrict__ in, const __bf16* __restrict__ wf,
    const float* __restrict__ bias, float* __restrict__ outp, int dir) {
  __shared__ v8bf lbs[16 * 64];                // 16KB
  int lane = threadIdx.x & 31;
  int wave = blockIdx.x * 16 + (threadIdx.x >> 5);
  int row0 = wave * 32;
  int c0base = (lane & 16) ? 8 : 0;
  const __bf16* arow[2];
  arow[0] = in + (size_t)(row0 + (lane & 15)) * MCH;
  arow[1] = arow[0] + 16 * MCH;

  {
    const v8bf* src = (const v8bf*)wf;
    #pragma unroll
    for (int i = 0; i < 2; ++i) {
      int idx = i * 512 + threadIdx.x;
      lbs[idx] = src[idx];
    }
  }
  __syncthreads();

  v8f acc[2][4];
  #pragma unroll
  for (int t2 = 0; t2 < 2; ++t2)
    #pragma unroll
    for (int i = 0; i < 4; ++i)
      #pragma unroll
      for (int j = 0; j < 8; ++j) acc[t2][i][j] = 0.0f;

  #pragma unroll
  for (int kt = 0; kt < 4; ++kt) {
    v16bf a0 = load_a(arow[0], kt * 32 + c0base);
    v16bf a1 = load_a(arow[1], kt * 32 + c0base);
    #pragma unroll
    for (int nt = 0; nt < 4; ++nt) {
      v16bf bb = load_b_lds(lbs, kt * 4 + nt, lane);
      acc[0][nt] = __builtin_amdgcn_wmma_f32_16x16x32_bf16(false, a0, false, bb, (short)0, acc[0][nt], false, false);
      acc[1][nt] = __builtin_amdgcn_wmma_f32_16x16x32_bf16(false, a1, false, bb, (short)0, acc[1][nt], false, false);
    }
  }

  int col = lane & 15;
  int radd = (lane & 16) ? 8 : 0;
  #pragma unroll
  for (int t2 = 0; t2 < 2; ++t2)
    #pragma unroll
    for (int nt = 0; nt < 4; ++nt) {
      int ch = nt * 16 + col;
      float bn = bias[ch];
      #pragma unroll
      for (int r = 0; r < 8; ++r) {
        int pr = row0 + t2 * 16 + r + radd;
        int bimg = pr / 225, yx = pr % 225;
        outp[((size_t)(bimg * 4 + dir) * 64 + ch) * 225 + yx] = acc[t2][nt][r] + bn;
      }
    }
}

// ---------------------------------------------------------------------------
extern "C" void kernel_launch(void* const* d_in, const int* in_sizes, int n_in,
                              void* d_out, int out_size, void* d_ws, size_t ws_size,
                              hipStream_t stream) {
  (void)in_sizes; (void)n_in; (void)out_size; (void)ws_size;
  const float* x    = (const float*)d_in[0];
  const float* w_d0 = (const float*)d_in[1];
  const float* b_d0 = (const float*)d_in[2];
  const float* w_dc = (const float*)d_in[3];
  const float* b_dc = (const float*)d_in[4];
  const float* w_px = (const float*)d_in[5];
  const float* b_px = (const float*)d_in[6];
  const float* w_c1 = (const float*)d_in[7];
  const float* b_c1 = (const float*)d_in[8];
  const float* w_c2 = (const float*)d_in[9];
  const float* b_c2 = (const float*)d_in[10];
  const float* w_f  = (const float*)d_in[11];
  const float* b_f  = (const float*)d_in[12];
  float* out = (float*)d_out;

  // workspace carve-up (bf16 elements)
  __bf16* cur = (__bf16*)d_ws;                       // NPIX*128
  __bf16* tmp = cur + (size_t)NPIX * MCH;            // NPIX*128
  __bf16* fdc = tmp + (size_t)NPIX * MCH;            // 12 mats * 32 frags * 512
  __bf16* fpx = fdc + 12 * 32 * 512;                 // 4  * 32 * 512
  __bf16* fc1 = fpx + 4 * 32 * 512;                  // 32 * 512
  __bf16* fc2 = fc1 + 32 * 512;                      // 32 * 512
  __bf16* ff  = fc2 + 32 * 512;                      // 16 * 512

  // pack weights into per-lane WMMA fragment order (tiny, runs once/launch)
  prep_kernel<<<768, 256, 0, stream>>>(w_dc, fdc, 12, 128, 128);
  prep_kernel<<<256, 256, 0, stream>>>(w_px, fpx, 4, 128, 128);
  prep_kernel<<<64,  256, 0, stream>>>(w_c1, fc1, 1, 128, 128);
  prep_kernel<<<64,  256, 0, stream>>>(w_c2, fc2, 1, 128, 128);
  prep_kernel<<<32,  256, 0, stream>>>(w_f,  ff,  1, 64, 128);

  // tap offsets per direction: (w0) and (w2); w1 is center
  static const int DY0[4] = {0, -1, -1,  1};
  static const int DX0[4] = {-1, 0, -1, -1};
  static const int DY2[4] = {0,  1,  1, -1};
  static const int DX2[4] = {1,  0,  1,  1};

  for (int d = 0; d < 4; ++d) {
    init_dirconv_kernel<<<(NPIX * MCH) / 256, 256, 0, stream>>>(
        x, w_d0, b_d0, cur, DY0[d], DX0[d], DY2[d], DX2[d]);
    for (int i = 0; i < 4; ++i) {
      dirconv_kernel<<<GBLK, 512, 0, stream>>>(
          cur, tmp, fdc + (size_t)i * 3 * 32 * 512, b_dc + i * MCH,
          DY0[d], DX0[d], DY2[d], DX2[d]);
      gemm1x1_kernel<<<GBLK, 512, 0, stream>>>(
          tmp, cur, fpx + (size_t)i * 32 * 512, b_px + i * MCH, cur, 1);
    }
    // Conv0dResBlock
    gemm1x1_kernel<<<GBLK, 512, 0, stream>>>(cur, tmp, fc1, b_c1, (const __bf16*)nullptr, 1);
    gemm1x1_kernel<<<GBLK, 512, 0, stream>>>(tmp, cur, fc2, b_c2, cur, 1);
    // final 1x1 -> f32 output slice for this direction
    final_kernel<<<GBLK, 512, 0, stream>>>(cur, ff, b_f, out, d);
  }
}
